// Renet_12970801234545
// MI455X (gfx1250) — compile-verified, matrix operand order checked by reading, not run on hardware
//
#include <hip/hip_runtime.h>
#include <hip/hip_bf16.h>
#include <stdint.h>

// ---------------------------------------------------------------------------
// ReNet (2x BiLSTM + 1x1 conv) for MI455X (gfx1250, wave32, WMMA).
// Per LSTM step: G[m, 512] = [x_t | h_{t-1}](f16) @ Wcat^T(f16), f32 accum via
// v_wmma_f32_16x16x32_f16; gates fused lane-locally in the C layout.
// Wave tile: 32 rows x 16 hidden x 4 gates = 8 WMMAs per 32-wide K chunk
// (2 A-frags + 4 B-frags -> 8 WMMAs), weight frags prefetched one K-step ahead.
// ---------------------------------------------------------------------------

typedef __attribute__((ext_vector_type(16))) _Float16 v16h;
typedef __attribute__((ext_vector_type(8)))  _Float16 h8;
typedef __attribute__((ext_vector_type(8)))  float    v8f;

union F16x16 { v16h v; h8 h[2]; };

#define B_     64
#define SIZE_  64
#define HID_   128
#define NSEQ_  4096   // B*SIZE

static __device__ __forceinline__ float sigf(float x) {
    return 1.0f / (1.0f + __expf(-x));
}
static __device__ __forceinline__ float tanh_fast(float x) {
    x = fminf(15.0f, fmaxf(-15.0f, x));
    float e = __expf(2.0f * x);
    return (e - 1.0f) / (e + 1.0f);
}

// ---------------------------------------------------------------------------
// Pack x [B,C,H,W] f32 -> Xv [ (b*64+h) ][ w ][ c ] f16   (rows for vertical LSTM)
// ---------------------------------------------------------------------------
__global__ void pack_x_kernel(const float* __restrict__ x, _Float16* __restrict__ dst) {
    long i = (long)blockIdx.x * blockDim.x + threadIdx.x;   // dst linear index
    if (i >= (long)B_ * 64 * 64 * 64) return;
    int c = i & 63;
    int w = (i >> 6) & 63;
    int h = (i >> 12) & 63;
    int b = (int)(i >> 18);
    float v = x[(((size_t)b * 64 + c) * 64 + h) * 64 + w];
    dst[i] = (_Float16)v;
}

// ---------------------------------------------------------------------------
// Pack LSTM weights (concat [w_ih | w_hh]) f32 -> fragment-ready f16:
//   dst[dir][ntile(32)][kt(K/32)][lane(32)][e(16)]
//   value = Wcat[dir][n][k],  n = ntile*16 + lane%16,
//   k = kt*32 + (e/8)*16 + (lane/16)*8 + (e%8)
// ---------------------------------------------------------------------------
__global__ void pack_w_kernel(const float* __restrict__ ih, const float* __restrict__ hh,
                              _Float16* __restrict__ dst, int K, int KX) {
    int KT = K / 32;
    long total = 2L * 32 * KT * 32 * 16;
    long i = (long)blockIdx.x * blockDim.x + threadIdx.x;
    if (i >= total) return;
    int e    = (int)(i & 15);
    int lane = (int)((i >> 4) & 31);
    long r   = i >> 9;
    int kt    = (int)(r % KT);  r /= KT;
    int ntile = (int)(r % 32);
    int dir   = (int)(r / 32);
    int n = ntile * 16 + (lane & 15);
    int k = kt * 32 + ((e >> 3) << 4) + ((lane >> 4) << 3) + (e & 7);
    float v = (k < KX) ? ih[((size_t)dir * 512 + n) * KX + k]
                       : hh[((size_t)dir * 512 + n) * HID_ + (k - KX)];
    dst[i] = (_Float16)v;
}

// conv_w [256][256] f32 -> dst[ntile(16)][kt(8)][lane(32)][e(16)] f16
__global__ void pack_convw_kernel(const float* __restrict__ w, _Float16* __restrict__ dst) {
    int i = blockIdx.x * blockDim.x + threadIdx.x;
    if (i >= 16 * 8 * 32 * 16) return;
    int e    = i & 15;
    int lane = (i >> 4) & 31;
    int kt   = (i >> 9) & 7;
    int ntile = i >> 12;
    int n = ntile * 16 + (lane & 15);
    int k = kt * 32 + ((e >> 3) << 4) + ((lane >> 4) << 3) + (e & 7);
    dst[i] = (_Float16)w[(size_t)n * 256 + k];
}

__global__ void zero_words_kernel(uint32_t* __restrict__ p, long n_words) {
    long i = (long)blockIdx.x * blockDim.x + threadIdx.x;
    if (i < n_words) p[i] = 0u;
}

// ---------------------------------------------------------------------------
// One fused LSTM time step, both directions (blockIdx.y = dir).
//   K  = KX + 128 (concat feature dim), KX = input features.
//   Xseq: [4096][64][KX] f16, sequence inputs
//   Hbuf/Cbuf: [2][4096][128] state (h in f16, c in f32)
//   Wpack: packed weights as above. bias: [2][512] f32 (i,f,g,o * 128)
//   outHS: f16 hidden-state output, layout selected by OUT_MODE:
//     OUT_MODE 0 (vertical):  ((b*64 + t_eff)*64 + h)*256 + dir*128 + j, n=(b,h)
//     OUT_MODE 1 (horizontal): (n*64 + t_eff)*256 + dir*128 + j,         n=(b,w)
// Workgroup: 256 thr = 8 waves; WG tile = 32 rows x 128 hidden.
// Wave = one 16-wide hidden tile, both 16-row tiles, all 4 gates:
// 8 WMMAs per 32-wide K chunk from 2 A-frags + 4 B-frags.
// ---------------------------------------------------------------------------
template <int K, int KX, int OUT_MODE>
__global__ __launch_bounds__(256)
void lstm_step_kernel(const _Float16* __restrict__ Xseq,
                      _Float16* __restrict__ Hbuf,
                      float* __restrict__ Cbuf,
                      const _Float16* __restrict__ Wpack,
                      const float* __restrict__ bias,
                      _Float16* __restrict__ outHS,
                      int t) {
    constexpr int KT = K / 32;
    __shared__ _Float16 Atile[32 * K];

    const int dir   = blockIdx.y;
    const int row0  = blockIdx.x * 32;
    const int t_eff = dir ? (63 - t) : t;
    const int tid   = threadIdx.x;

    // ---- stage A tile: rows row0..row0+31, features [x_t | h_{t-1}] ----
    constexpr int SEGS = K / 8;               // 8-halfword segments per row
    for (int i = tid; i < 32 * SEGS; i += 256) {   // exact multiple: no divergence
        int r = i / SEGS, s = i % SEGS;
        int n = row0 + r;
        h8 val;
        if (s * 8 < KX) {
            val = *(const h8*)(Xseq + ((size_t)n * 64 + t_eff) * KX + s * 8);
        } else {
            val = *(const h8*)(Hbuf + ((size_t)dir * NSEQ_ + n) * HID_ + (s * 8 - KX));
        }
        *(h8*)(&Atile[r * K + s * 8]) = val;
    }
    __syncthreads();

    const int lane = tid & 31;
    const int wave = tid >> 5;          // hidden tile 0..7
    const int lm = lane & 15;           // matrix row (A) / column (B,C)
    const int lh = lane >> 4;           // half-select

    v8f acc[2][4] = {};                 // [row tile][gate i,f,g,o]

    for (int kt = 0; kt < KT; ++kt) {
        F16x16 a[2];
#pragma unroll
        for (int rt = 0; rt < 2; ++rt) {
            const _Float16* arow = &Atile[(rt * 16 + lm) * K + kt * 32 + lh * 8];
            a[rt].h[0] = *(const h8*)(arow);
            a[rt].h[1] = *(const h8*)(arow + 16);
        }
#pragma unroll
        for (int g = 0; g < 4; ++g) {
            int ntile = g * 8 + wave;   // gate g, hidden tile `wave`
            const _Float16* bp = Wpack + ((((size_t)dir * 32 + ntile) * KT + kt) * 32 + lane) * 16;
            F16x16 b;
            b.h[0] = *(const h8*)(bp);
            b.h[1] = *(const h8*)(bp + 8);
            if (kt + 1 < KT)            // pull next K-step's fragment toward L1
                __builtin_prefetch(bp + 32 * 16, 0, 3);
            acc[0][g] = __builtin_amdgcn_wmma_f32_16x16x32_f16(
                false, a[0].v, false, b.v, (short)0, acc[0][g], false, false);
            acc[1][g] = __builtin_amdgcn_wmma_f32_16x16x32_f16(
                false, a[1].v, false, b.v, (short)0, acc[1][g], false, false);
        }
    }

    // ---- fused gate math: lane owns column j for 2x8 rows ----
    const int j  = wave * 16 + lm;      // hidden index
    const float bi = bias[dir * 512 + 0 * HID_ + j];
    const float bf = bias[dir * 512 + 1 * HID_ + j];
    const float bg = bias[dir * 512 + 2 * HID_ + j];
    const float bo = bias[dir * 512 + 3 * HID_ + j];

#pragma unroll
    for (int rt = 0; rt < 2; ++rt) {
#pragma unroll
        for (int r = 0; r < 8; ++r) {
            int m = rt * 16 + r + 8 * lh;   // C-matrix row within WG tile
            int n = row0 + m;
            size_t sidx = ((size_t)dir * NSEQ_ + n) * HID_ + j;
            float i_ = sigf(acc[rt][0][r] + bi);
            float f_ = sigf(acc[rt][1][r] + bf);
            float g_ = tanh_fast(acc[rt][2][r] + bg);
            float o_ = sigf(acc[rt][3][r] + bo);
            float c  = f_ * Cbuf[sidx] + i_ * g_;
            Cbuf[sidx] = c;
            float hval = o_ * tanh_fast(c);
            Hbuf[sidx] = (_Float16)hval;

            size_t oidx;
            if (OUT_MODE == 0) {
                int b  = n >> 6, hh_ = n & 63;
                oidx = (((size_t)b * 64 + t_eff) * 64 + hh_) * 256 + (size_t)dir * HID_ + j;
            } else {
                oidx = ((size_t)n * 64 + t_eff) * 256 + (size_t)dir * HID_ + j;
            }
            outHS[oidx] = (_Float16)hval;
        }
    }
}

// ---------------------------------------------------------------------------
// 1x1 conv as WMMA GEMM: out[b,o,h,w] = sum_ch H2[(b,w),h,ch]*W[o,ch] + cb[o]
// M = 262144 rows (r = (b*64+w)*64+h), K = 256, N = 256.
// WG: 256 thr = 8 waves; WG tile = 32 rows x 256 cols.
// Wave covers 32 out channels (2 ntiles) x 2 row tiles: 4 WMMAs per K chunk.
// ---------------------------------------------------------------------------
__global__ __launch_bounds__(256)
void conv1x1_kernel(const _Float16* __restrict__ H2,
                    const _Float16* __restrict__ Wp,    // [16][8][32][16]
                    const float* __restrict__ cb,
                    float* __restrict__ out) {
    __shared__ _Float16 Atile[32 * 256];
    const int row0 = blockIdx.x * 32;
    const int tid  = threadIdx.x;

    for (int i = tid; i < 32 * 32; i += 256) {   // 32 x 8-halfword segments/row
        int r = i >> 5, s = i & 31;
        *(h8*)(&Atile[r * 256 + s * 8]) =
            *(const h8*)(H2 + (size_t)(row0 + r) * 256 + s * 8);
    }
    __syncthreads();

    const int lane = tid & 31;
    const int wave = tid >> 5;
    const int lm = lane & 15, lh = lane >> 4;

    v8f acc[2][2] = {};                 // [row tile][ntile-of-wave]
    for (int kt = 0; kt < 8; ++kt) {
        F16x16 a[2];
#pragma unroll
        for (int rt = 0; rt < 2; ++rt) {
            const _Float16* arow = &Atile[(rt * 16 + lm) * 256 + kt * 32 + lh * 8];
            a[rt].h[0] = *(const h8*)(arow);
            a[rt].h[1] = *(const h8*)(arow + 16);
        }
#pragma unroll
        for (int g = 0; g < 2; ++g) {
            int ntile = wave * 2 + g;
            const _Float16* bp = Wp + (((size_t)ntile * 8 + kt) * 32 + lane) * 16;
            F16x16 b;
            b.h[0] = *(const h8*)(bp);
            b.h[1] = *(const h8*)(bp + 8);
            if (kt < 7)
                __builtin_prefetch(bp + 32 * 16, 0, 3);
            acc[0][g] = __builtin_amdgcn_wmma_f32_16x16x32_f16(
                false, a[0].v, false, b.v, (short)0, acc[0][g], false, false);
            acc[1][g] = __builtin_amdgcn_wmma_f32_16x16x32_f16(
                false, a[1].v, false, b.v, (short)0, acc[1][g], false, false);
        }
    }

#pragma unroll
    for (int rt = 0; rt < 2; ++rt) {
#pragma unroll
        for (int g = 0; g < 2; ++g) {
            int o = (wave * 2 + g) * 16 + lm;
            float bv = cb[o];
#pragma unroll
            for (int r = 0; r < 8; ++r) {
                int m = rt * 16 + r + 8 * lh;
                int rg = row0 + m;          // rg = (b*64+w)*64 + h
                int h  = rg & 63;
                int bw = rg >> 6;
                int w  = bw & 63;
                int b  = bw >> 6;
                out[(((size_t)b * 256 + o) << 12) + (h << 6) + w] = acc[rt][g][r] + bv;
            }
        }
    }
}

// ---------------------------------------------------------------------------
// Host side
// ---------------------------------------------------------------------------
extern "C" void kernel_launch(void* const* d_in, const int* in_sizes, int n_in,
                              void* d_out, int out_size, void* d_ws, size_t ws_size,
                              hipStream_t stream) {
    (void)in_sizes; (void)n_in; (void)out_size; (void)ws_size;

    const float* x      = (const float*)d_in[0];   // [64,64,64,64]
    const float* vw_ih  = (const float*)d_in[1];   // [2,512,64]
    const float* vw_hh  = (const float*)d_in[2];   // [2,512,128]
    const float* vb     = (const float*)d_in[3];   // [2,512]
    const float* hw_ih  = (const float*)d_in[4];   // [2,512,256]
    const float* hw_hh  = (const float*)d_in[5];   // [2,512,128]
    const float* hb     = (const float*)d_in[6];   // [2,512]
    const float* conv_w = (const float*)d_in[7];   // [256,256]
    const float* conv_b = (const float*)d_in[8];   // [256]
    float* out = (float*)d_out;

    // ---- workspace partition (all 256B aligned) ----
    char* ws = (char*)d_ws;
    size_t off = 0;
    auto take = [&](size_t bytes) {
        size_t o = off;
        off = (off + bytes + 255) & ~(size_t)255;
        return o;
    };
    _Float16* XV  = (_Float16*)(ws + take((size_t)NSEQ_ * 64 * 64 * 2));      //  32 MB
    _Float16* VH  = (_Float16*)(ws + take((size_t)NSEQ_ * 64 * 256 * 2));     // 128 MB
    _Float16* H2  = (_Float16*)(ws + take((size_t)NSEQ_ * 64 * 256 * 2));     // 128 MB
    _Float16* WPV = (_Float16*)(ws + take((size_t)2 * 32 * 6 * 32 * 16 * 2));
    _Float16* WPH = (_Float16*)(ws + take((size_t)2 * 32 * 12 * 32 * 16 * 2));
    _Float16* WPC = (_Float16*)(ws + take((size_t)16 * 8 * 32 * 16 * 2));
    char*     HC  = ws + take((size_t)2 * NSEQ_ * HID_ * 2 +
                              (size_t)2 * NSEQ_ * HID_ * 4);   // Hbuf(f16)+Cbuf(f32), contiguous
    _Float16* HB  = (_Float16*)HC;
    float*    CB  = (float*)(HC + (size_t)2 * NSEQ_ * HID_ * 2);
    const long hc_words = ((size_t)2 * NSEQ_ * HID_ * 2 + (size_t)2 * NSEQ_ * HID_ * 4) / 4;

    // ---- packing (cheap, every call for determinism) ----
    {
        long n = (long)B_ * 64 * 64 * 64;
        pack_x_kernel<<<(unsigned)((n + 255) / 256), 256, 0, stream>>>(x, XV);
    }
    {
        long n = 2L * 32 * 6 * 32 * 16;
        pack_w_kernel<<<(unsigned)((n + 255) / 256), 256, 0, stream>>>(vw_ih, vw_hh, WPV, 192, 64);
    }
    {
        long n = 2L * 32 * 12 * 32 * 16;
        pack_w_kernel<<<(unsigned)((n + 255) / 256), 256, 0, stream>>>(hw_ih, hw_hh, WPH, 384, 256);
    }
    pack_convw_kernel<<<(16 * 8 * 32 * 16) / 256, 256, 0, stream>>>(conv_w, WPC);

    dim3 stepGrid(NSEQ_ / 32, 2);   // 128 row-tiles x 2 directions

    // ---- vertical BiLSTM: seq over w, K = 64 + 128 ----
    zero_words_kernel<<<(unsigned)((hc_words + 255) / 256), 256, 0, stream>>>((uint32_t*)HC, hc_words);
    for (int t = 0; t < 64; ++t) {
        lstm_step_kernel<192, 64, 0><<<stepGrid, 256, 0, stream>>>(XV, HB, CB, WPV, vb, VH, t);
    }

    // ---- horizontal BiLSTM: seq over h, K = 256 + 128 ----
    zero_words_kernel<<<(unsigned)((hc_words + 255) / 256), 256, 0, stream>>>((uint32_t*)HC, hc_words);
    for (int t = 0; t < 64; ++t) {
        lstm_step_kernel<384, 256, 1><<<stepGrid, 256, 0, stream>>>(VH, HB, CB, WPH, hb, H2, t);
    }

    // ---- 1x1 conv GEMM ----
    conv1x1_kernel<<<(NSEQ_ * 64) / 32, 256, 0, stream>>>(H2, WPC, conv_b, out);
}
